// Attention1d_19232863551966
// MI455X (gfx1250) — matrix operand order repaired and checked by reading, not compiled, not run on hardware
//
#include <hip/hip_runtime.h>
#include <cstdint>

typedef __attribute__((ext_vector_type(8)))  _Float16 v8h;
typedef __attribute__((ext_vector_type(16))) _Float16 v16h;
typedef __attribute__((ext_vector_type(8)))  float    v8f;

constexpr int B_ = 16;
constexpr int N_ = 8192;
constexpr int D_ = 64;
constexpr int CHUNKS = 16;                     // N-split for phase 1
constexpr int ROWS_PER_CHUNK = N_ / CHUNKS;    // 512
constexpr int ROWS_PER_BLOCK = 128;            // phase 2 row tile per block
constexpr int LDK = 40;   // LDS row stride (halfwords) for 32-wide tiles: 80B, 16B aligned
constexpr int LDE = 72;   // LDS row stride (halfwords) for 64-wide kvT rows: 144B, 16B aligned

__device__ __forceinline__ float phi(float x) {
    // elu(x)+1 == x+1 (x>0), exp(x) (x<=0)
    return x > 0.0f ? x + 1.0f : __expf(x);
}

__device__ __forceinline__ v16h cat8(v8h a, v8h b) {
    return __builtin_shufflevector(a, b, 0,1,2,3,4,5,6,7,8,9,10,11,12,13,14,15);
}

// ---------------------------------------------------------------------------
// Phase 1: partial KV[b] = sum_n phi(k[b,n,:])^T v[b,n,:] over this block's
// N-chunk. 16 waves each own one 16x16 tile of the 64x64 result.
// ---------------------------------------------------------------------------
template <bool ATOMIC>
__global__ __launch_bounds__(512)
void kv_kernel(const float* __restrict__ k, const float* __restrict__ v,
               float* __restrict__ dst) {
    __shared__ _Float16 kT[D_ * LDK];   // kT[d][n_local], phi applied
    __shared__ _Float16 vT[D_ * LDK];   // vT[e][n_local]

    const int b     = blockIdx.x / CHUNKS;
    const int chunk = blockIdx.x % CHUNKS;
    const int t  = threadIdx.x;
    const int ln = t & 31;
    const int w  = t >> 5;          // wave 0..15
    const int dt = w >> 2;          // A-tile row block (d)
    const int et = w & 3;           // B-tile col block (e)
    const int half = ln >> 4;
    const int mrow = ln & 15;
    const int s_row = t >> 4;       // staging: n_local 0..31
    const int s_col = (t & 15) * 4; // staging: d/e 0..60 step 4

    const size_t base = ((size_t)b * N_ + (size_t)chunk * ROWS_PER_CHUNK) * D_;
    v8f acc = {};

    for (int step = 0; step < ROWS_PER_CHUNK / 32; ++step) {
        const size_t off = base + (size_t)(step * 32 + s_row) * D_ + s_col;
        const float4 kq = *(const float4*)(k + off);
        const float4 vq = *(const float4*)(v + off);
        __syncthreads();   // previous iteration's LDS reads complete
        kT[(s_col + 0) * LDK + s_row] = (_Float16)phi(kq.x);
        kT[(s_col + 1) * LDK + s_row] = (_Float16)phi(kq.y);
        kT[(s_col + 2) * LDK + s_row] = (_Float16)phi(kq.z);
        kT[(s_col + 3) * LDK + s_row] = (_Float16)phi(kq.w);
        vT[(s_col + 0) * LDK + s_row] = (_Float16)vq.x;
        vT[(s_col + 1) * LDK + s_row] = (_Float16)vq.y;
        vT[(s_col + 2) * LDK + s_row] = (_Float16)vq.z;
        vT[(s_col + 3) * LDK + s_row] = (_Float16)vq.w;
        __syncthreads();

        // A fragment 16x32 (M=d, K=n): elems 0..7 -> K = 8*half + i,
        // elems 8..15 -> K = 16 + 8*half + (i-8)   (contiguous 16B runs)
        const _Float16* arow = &kT[(dt * 16 + mrow) * LDK];
        v16h afrag = cat8(*(const v8h*)(arow + 8 * half),
                          *(const v8h*)(arow + 16 + 8 * half));
        // B fragment 32x16 (K=n, N=e): elems 0..7 -> K = 8*half + i,
        // elems 8..15 -> K = 16 + 8*half + (i-8)
        const _Float16* brow = &vT[(et * 16 + mrow) * LDK];
        v16h bfrag = cat8(*(const v8h*)(brow + 8 * half),
                          *(const v8h*)(brow + 16 + 8 * half));

        acc = __builtin_amdgcn_wmma_f32_16x16x32_f16(false, afrag, false, bfrag,
                                                     (short)0, acc, false, false);
    }

    // C layout: VGPR r, lane ln -> M = r + 8*half, N = ln & 15
    float* outp = ATOMIC ? dst + (size_t)b * D_ * D_
                         : dst + ((size_t)b * CHUNKS + chunk) * D_ * D_;
#pragma unroll
    for (int r = 0; r < 8; ++r) {
        const int gm = dt * 16 + r + 8 * half;
        const int gn = et * 16 + mrow;
        if (ATOMIC) atomicAdd(&outp[gm * D_ + gn], acc[r]);
        else        outp[gm * D_ + gn] = acc[r];
    }
}

__global__ void reduce_kv(const float* __restrict__ partial, float* __restrict__ kv) {
    const int i = blockIdx.x * blockDim.x + threadIdx.x;   // over B_*64*64
    const int b   = i >> 12;
    const int off = i & 4095;
    float s = 0.0f;
#pragma unroll
    for (int c = 0; c < CHUNKS; ++c)
        s += partial[((size_t)b * CHUNKS + c) * (D_ * D_) + off];
    kv[i] = s;
}

__global__ void zero_f32(float* p, int n) {
    const int i = blockIdx.x * blockDim.x + threadIdx.x;
    if (i < n) p[i] = 0.0f;
}

// ---------------------------------------------------------------------------
// Phase 2: out[b, n, :] = phi(q[b, n, :]) @ KV[b] * (1/(N*D))
// 128 rows per block; each wave: one 16-row tile x two 16-col tiles.
// ---------------------------------------------------------------------------
__global__ __launch_bounds__(512)
void out_kernel(const float* __restrict__ q, const float* __restrict__ kv,
                float* __restrict__ out) {
    __shared__ _Float16 kvT[D_ * LDE];   // kvT[e][d]

    const int b  = blockIdx.x / (N_ / ROWS_PER_BLOCK);
    const int nb = blockIdx.x % (N_ / ROWS_PER_BLOCK);
    const int n_base = nb * ROWS_PER_BLOCK;
    const int t  = threadIdx.x;
    const int ln = t & 31;
    const int w  = t >> 5;
    const int nt = w & 7;            // row tile 0..7
    const int eg = (w >> 3) * 2;     // col tiles eg, eg+1
    const int half = ln >> 4;
    const int mrow = ln & 15;

    // stage KV transposed, f32 -> f16
    const float* kvb = kv + (size_t)b * D_ * D_;
#pragma unroll
    for (int it = 0; it < 2; ++it) {
        const int idx = t + it * 512;      // float4 index 0..1023
        const int d = idx >> 4;
        const int e = (idx & 15) * 4;
        const float4 kq = *(const float4*)(kvb + d * D_ + e);
        kvT[(e + 0) * LDE + d] = (_Float16)kq.x;
        kvT[(e + 1) * LDE + d] = (_Float16)kq.y;
        kvT[(e + 2) * LDE + d] = (_Float16)kq.z;
        kvT[(e + 3) * LDE + d] = (_Float16)kq.w;
    }
    __syncthreads();

    const int nrow = n_base + nt * 16 + mrow;
    const float* qrow = q + ((size_t)b * N_ + nrow) * D_;
    v8f acc0 = {}, acc1 = {};
#pragma unroll
    for (int kk = 0; kk < 2; ++kk) {
        const int koff = kk * 32;
        // A fragment straight from global: K-runs are contiguous in q's row
        const float4 qa = *(const float4*)(qrow + koff + 8 * half);
        const float4 qb = *(const float4*)(qrow + koff + 8 * half + 4);
        const float4 qc = *(const float4*)(qrow + koff + 16 + 8 * half);
        const float4 qd = *(const float4*)(qrow + koff + 16 + 8 * half + 4);
        v16h a;
        a[0]=(_Float16)phi(qa.x);  a[1]=(_Float16)phi(qa.y);
        a[2]=(_Float16)phi(qa.z);  a[3]=(_Float16)phi(qa.w);
        a[4]=(_Float16)phi(qb.x);  a[5]=(_Float16)phi(qb.y);
        a[6]=(_Float16)phi(qb.z);  a[7]=(_Float16)phi(qb.w);
        a[8]=(_Float16)phi(qc.x);  a[9]=(_Float16)phi(qc.y);
        a[10]=(_Float16)phi(qc.z); a[11]=(_Float16)phi(qc.w);
        a[12]=(_Float16)phi(qd.x); a[13]=(_Float16)phi(qd.y);
        a[14]=(_Float16)phi(qd.z); a[15]=(_Float16)phi(qd.w);

        const _Float16* b0 = &kvT[(eg * 16 + mrow) * LDE + koff + 8 * half];
        const _Float16* b1 = &kvT[((eg + 1) * 16 + mrow) * LDE + koff + 8 * half];
        v16h bf0 = cat8(*(const v8h*)(b0), *(const v8h*)(b0 + 16));
        v16h bf1 = cat8(*(const v8h*)(b1), *(const v8h*)(b1 + 16));

        acc0 = __builtin_amdgcn_wmma_f32_16x16x32_f16(false, a, false, bf0,
                                                      (short)0, acc0, false, false);
        acc1 = __builtin_amdgcn_wmma_f32_16x16x32_f16(false, a, false, bf1,
                                                      (short)0, acc1, false, false);
    }

    const float scale = 1.0f / (float)((size_t)D_ * N_);
    float* outb = out + (size_t)b * N_ * D_;
#pragma unroll
    for (int r = 0; r < 8; ++r) {
        const int gm = n_base + nt * 16 + r + 8 * half;
        outb[(size_t)gm * D_ + eg * 16 + mrow]       = acc0[r] * scale;
        outb[(size_t)gm * D_ + (eg + 1) * 16 + mrow] = acc1[r] * scale;
    }
}

extern "C" void kernel_launch(void* const* d_in, const int* in_sizes, int n_in,
                              void* d_out, int out_size, void* d_ws, size_t ws_size,
                              hipStream_t stream) {
    const float* q = (const float*)d_in[0];
    const float* k = (const float*)d_in[1];
    const float* v = (const float*)d_in[2];
    float* out = (float*)d_out;

    float* kv_final = (float*)d_ws;                          // B*64*64 f32
    float* partial  = kv_final + (size_t)B_ * D_ * D_;       // B*CHUNKS*64*64 f32
    const size_t need_partial =
        (size_t)B_ * D_ * D_ * (size_t)(CHUNKS + 1) * sizeof(float);

    if (ws_size >= need_partial) {
        // deterministic path: per-chunk partials + serial sum over chunks
        kv_kernel<false><<<B_ * CHUNKS, 512, 0, stream>>>(k, v, partial);
        reduce_kv<<<(B_ * D_ * D_) / 256, 256, 0, stream>>>(partial, kv_final);
    } else {
        // small-workspace fallback: zero + float atomics
        const int n = B_ * D_ * D_;
        zero_f32<<<(n + 255) / 256, 256, 0, stream>>>(kv_final, n);
        kv_kernel<true><<<B_ * CHUNKS, 512, 0, stream>>>(k, v, kv_final);
    }

    out_kernel<<<B_ * (N_ / ROWS_PER_BLOCK), 512, 0, stream>>>(q, kv_final, out);
}
